// EvRaindrop_v1_72550587564658
// MI455X (gfx1250) — matrix operand order repaired and checked by reading, not compiled, not run on hardware
//
#include <hip/hip_runtime.h>
#include <hip/hip_bf16.h>
#include <math.h>

// ---------------- problem constants ----------------
#define TT     256   // time steps
#define BB     128   // batch
#define DINP   256   // core nodes
#define DSTAT  256   // static nodes
#define DD     512   // per-node feature width (T*d_ob)
#define KK     6     // top-k
#define INV_SQRT_D 0.044194173824159216f  // 1/sqrt(512)

typedef __attribute__((ext_vector_type(16))) _Float16 v16h;
typedef __attribute__((ext_vector_type(8)))  float    v8f;

union Frag16 { v16h v; uint4 q[2]; };

#define PSTR 520   // LDS row stride (halfs) for P tile: 1040B = 65*16 -> 16B aligned, bank-skewed
#define LSTR 260   // LDS row stride (floats) for L tile

// ---------------------------------------------------------------------------
// K0: X[b][c][2t+ob] = relu(src[t,b,c])  (f16, duplicated pairs), LDS transpose
// ---------------------------------------------------------------------------
__global__ __launch_bounds__(256) void k_prep(const float* __restrict__ src,
                                              _Float16* __restrict__ X) {
  __shared__ float tile[32][33];
  const int lx = threadIdx.x & 31, ly = threadIdx.x >> 5;  // 32 x 8
  const int b  = blockIdx.x >> 6;
  const int rem = blockIdx.x & 63;
  const int c0 = (rem & 7) * 32, t0 = (rem >> 3) * 32;
  #pragma unroll
  for (int i = 0; i < 4; ++i) {
    const int t = t0 + ly + i * 8;
    const float v = src[((size_t)t * BB + b) * DINP + c0 + lx];
    tile[lx][ly + i * 8] = fmaxf(v, 0.f);          // tile[c'][t']
  }
  __syncthreads();
  #pragma unroll
  for (int i = 0; i < 4; ++i) {
    const int cc = ly + i * 8;
    const _Float16 hv = (_Float16)tile[cc][lx];
    union { _Float16 h[2]; unsigned int w; } p; p.h[0] = hv; p.h[1] = hv;
    *(unsigned int*)&X[((size_t)(b * DINP + c0 + cc)) * DD + 2 * (t0 + lx)] = p.w;
  }
}

// ---------------------------------------------------------------------------
// K1: column sums + chained matvecs: ksum/vsum/wsum, u = Wq@ksum, w2 = Wv@u
// ---------------------------------------------------------------------------
__global__ __launch_bounds__(512) void k_weights(const float* __restrict__ Wq,
                                                 const float* __restrict__ Wk,
                                                 const float* __restrict__ Wv,
                                                 const float* __restrict__ Wc,
                                                 float* __restrict__ ksum,
                                                 float* __restrict__ vsum,
                                                 float* __restrict__ wsum,
                                                 float* __restrict__ w2Out) {
  __shared__ float sks[512], suu[512];
  const int t = threadIdx.x;
  float sk = 0.f, sv = 0.f;
  for (int l = 0; l < DD; ++l) { sk += Wk[(size_t)l * DD + t]; sv += Wv[(size_t)l * DD + t]; }
  sks[t] = sk; ksum[t] = sk; vsum[t] = sv;
  if (t < DINP) { float sw = 0.f; for (int l = 0; l < DD; ++l) sw += Wc[(size_t)t * DD + l]; wsum[t] = sw; }
  __syncthreads();
  float su = 0.f;
  for (int d = 0; d < DD; ++d) su += Wq[(size_t)t * DD + d] * sks[d];
  suu[t] = su;
  __syncthreads();
  float s2 = 0.f;
  for (int d = 0; d < DD; ++d) s2 += Wv[(size_t)t * DD + d] * suu[d];
  w2Out[t] = s2;
}

// ---------------------------------------------------------------------------
// K2: Mt[n][k] = inv_sqrt_d * sum_d Wq[k,d]*Wk[n,d]   (f16, transposed store)
// ---------------------------------------------------------------------------
__global__ __launch_bounds__(256) void k_mt(const float* __restrict__ Wq,
                                            const float* __restrict__ Wk,
                                            _Float16* __restrict__ Mt) {
  __shared__ float aq[16][17], ak[16][17];
  const int tx = threadIdx.x & 15, ty = threadIdx.x >> 4;
  const int k0 = (blockIdx.x & 31) * 16, n0 = (blockIdx.x >> 5) * 16;
  float acc = 0.f;
  for (int d0 = 0; d0 < DD; d0 += 16) {
    aq[ty][tx] = Wq[(size_t)(k0 + ty) * DD + d0 + tx];
    ak[ty][tx] = Wk[(size_t)(n0 + ty) * DD + d0 + tx];
    __syncthreads();
    #pragma unroll
    for (int dd = 0; dd < 16; ++dd) acc += aq[tx][dd] * ak[ty][dd];
    __syncthreads();
  }
  Mt[(size_t)(n0 + ty) * DD + (k0 + tx)] = (_Float16)(acc * INV_SQRT_D);
}

// ---------------------------------------------------------------------------
// K3: a[b,o] = relu(static[b,:] @ W_si[o,:] + b_si[o])
// ---------------------------------------------------------------------------
__global__ __launch_bounds__(256) void k_abs(const float* __restrict__ Wsi,
                                             const float* __restrict__ bsi,
                                             const float* __restrict__ stat,
                                             float* __restrict__ a) {
  const int idx = blockIdx.x * 256 + threadIdx.x;
  const int b = idx >> 8, o = idx & 255;
  float s = bsi[o];
  for (int q = 0; q < DSTAT; ++q) s += Wsi[(size_t)o * DSTAT + q] * stat[(size_t)b * DSTAT + q];
  a[idx] = fmaxf(s, 0.f);
}

// ---------------------------------------------------------------------------
// K4: logZ[b,s] = logsumexp_c( a[b,s]*wsum[c] + b_c[c] )
// ---------------------------------------------------------------------------
__global__ __launch_bounds__(256) void k_logz(const float* __restrict__ a,
                                              const float* __restrict__ wsum,
                                              const float* __restrict__ bc,
                                              float* __restrict__ logZ) {
  __shared__ float ws[256], bcs[256];
  const int s = threadIdx.x, b = blockIdx.x;
  ws[s] = wsum[s]; bcs[s] = bc[s];
  __syncthreads();
  const float av = a[b * 256 + s];
  float m = -1e30f;
  for (int c = 0; c < DINP; ++c) m = fmaxf(m, av * ws[c] + bcs[c]);
  float se = 0.f;
  for (int c = 0; c < DINP; ++c) se += expf(av * ws[c] + bcs[c] - m);
  logZ[b * 256 + s] = m + logf(se);
}

// ---------------------------------------------------------------------------
// K5: y[b,j] = X[b,j,:] . w2
// ---------------------------------------------------------------------------
__global__ __launch_bounds__(256) void k_y(const _Float16* __restrict__ X,
                                           const float* __restrict__ w2,
                                           float* __restrict__ y) {
  __shared__ float w2s[512];
  const int j = threadIdx.x, b = blockIdx.x;
  w2s[j] = w2[j]; w2s[j + 256] = w2[j + 256];
  __syncthreads();
  const _Float16* xr = X + ((size_t)b * DINP + j) * DD;
  float s = 0.f;
  for (int l = 0; l < DD; ++l) s += (float)xr[l] * w2s[l];
  y[b * 256 + j] = s;
}

// ---------------------------------------------------------------------------
// K6: per (b, 16-row tile): P = Xtile @ M  (WMMA, f16 in LDS),
//     L = P @ Xb^T (WMMA), fused softmax_j + dot with y -> s[b,i]
//     One wave / block (EXEC all ones as WMMA requires).
// ---------------------------------------------------------------------------
__global__ __launch_bounds__(32) void k_attn(const _Float16* __restrict__ X,
                                             const _Float16* __restrict__ Mt,
                                             const float* __restrict__ y,
                                             float* __restrict__ sOut) {
  __shared__ _Float16 Psh[16 * PSTR];
  __shared__ float    Lsh[16 * LSTR];
  const int lane = threadIdx.x;
  const int b = blockIdx.x >> 4;
  const int i0 = (blockIdx.x & 15) * 16;
  const int r = lane & 15, hi = lane >> 4;
  const _Float16* Xb = X + (size_t)b * DINP * DD;

  // ---- GEMM1: P[16x512] = Xtile[16x512] @ M[512x512] (M stored transposed) ----
  for (int nt = 0; nt < 32; ++nt) {
    const int n0 = nt * 16;
    v8f acc = {};
    for (int kt = 0; kt < 16; ++kt) {
      const int k0 = kt * 32;
      Frag16 af, bf;
      const _Float16* ap = Xb + (size_t)(i0 + r) * DD + k0 + hi * 8;   // A: k = hi*8+(e&7)+16*(e>>3)
      af.q[0] = *(const uint4*)(ap);
      af.q[1] = *(const uint4*)(ap + 16);
      const _Float16* bp = Mt + (size_t)(n0 + r) * DD + k0 + hi * 8;   // B: k = hi*8+(e&7)+16*(e>>3), n = lane&15
      bf.q[0] = *(const uint4*)(bp);
      bf.q[1] = *(const uint4*)(bp + 16);
      acc = __builtin_amdgcn_wmma_f32_16x16x32_f16(false, af.v, false, bf.v,
                                                   (short)0, acc, false, false);
    }
    #pragma unroll
    for (int v = 0; v < 8; ++v)
      Psh[(hi * 8 + v) * PSTR + n0 + r] = (_Float16)acc[v];            // D: row hi*8+v, col lane&15
  }
  __syncthreads();

  // ---- GEMM2: L[16x256] = P[16x512] @ Xb^T  (B rows read row-major in k) ----
  for (int jt = 0; jt < 16; ++jt) {
    const int j0 = jt * 16;
    v8f acc = {};
    for (int kt = 0; kt < 16; ++kt) {
      const int k0 = kt * 32;
      Frag16 af, bf;
      const _Float16* ap = &Psh[r * PSTR + k0 + hi * 8];
      af.q[0] = *(const uint4*)(ap);
      af.q[1] = *(const uint4*)(ap + 16);
      const _Float16* bp = Xb + (size_t)(j0 + r) * DD + k0 + hi * 8;
      __builtin_prefetch(bp + 32, 0, 1);                               // global_prefetch_b8
      bf.q[0] = *(const uint4*)(bp);
      bf.q[1] = *(const uint4*)(bp + 16);
      acc = __builtin_amdgcn_wmma_f32_16x16x32_f16(false, af.v, false, bf.v,
                                                   (short)0, acc, false, false);
    }
    #pragma unroll
    for (int v = 0; v < 8; ++v)
      Lsh[(hi * 8 + v) * LSTR + j0 + r] = acc[v];
  }
  __syncthreads();

  // ---- softmax over j (256) + dot with y: two lanes per row ----
  {
    const int row = lane >> 1, h = lane & 1;
    const float* Lr = &Lsh[row * LSTR + h * 128];
    float m = -1e30f;
    for (int j = 0; j < 128; ++j) m = fmaxf(m, Lr[j]);
    m = fmaxf(m, __shfl_xor(m, 1, 32));
    const float* yb = y + b * 256 + h * 128;
    float se = 0.f, sy = 0.f;
    for (int j = 0; j < 128; ++j) { const float e = expf(Lr[j] - m); se += e; sy += e * yb[j]; }
    se += __shfl_xor(se, 1, 32);
    sy += __shfl_xor(sy, 1, 32);
    if (h == 0) sOut[b * 256 + i0 + row] = sy / se;
  }
}

// ---------------------------------------------------------------------------
// K7: per (b,c): top-6 over s of a[b,s]*wsum[c]-logZ[b,s] (ties -> lower s),
//     then coeff[b,c] = sum_n softmax(s_bc*a_idx*inv_sqrt_d)_n * a_idx_n
// ---------------------------------------------------------------------------
__global__ __launch_bounds__(256) void k_coeff(const float* __restrict__ a,
                                               const float* __restrict__ logZ,
                                               const float* __restrict__ wsum,
                                               const float* __restrict__ sIn,
                                               float* __restrict__ coeff) {
  __shared__ float as[256], lz[256];
  const int c = threadIdx.x, b = blockIdx.x;
  as[c] = a[b * 256 + c]; lz[c] = logZ[b * 256 + c];
  __syncthreads();
  const float wc = wsum[c];
  float tv[KK]; int ti[KK];
  #pragma unroll
  for (int n = 0; n < KK; ++n) { tv[n] = -1e30f; ti[n] = 0; }
  for (int s = 0; s < DSTAT; ++s) {
    const float key = as[s] * wc - lz[s];
    if (key > tv[KK - 1]) {                    // strict >: ties keep lower index
      int p = KK - 1;
      while (p > 0 && key > tv[p - 1]) { tv[p] = tv[p - 1]; ti[p] = ti[p - 1]; --p; }
      tv[p] = key; ti[p] = s;
    }
  }
  const float sb = sIn[b * 256 + c];
  float lg[KK], m = -1e30f;
  #pragma unroll
  for (int n = 0; n < KK; ++n) { lg[n] = sb * as[ti[n]] * INV_SQRT_D; m = fmaxf(m, lg[n]); }
  float se = 0.f, num = 0.f;
  #pragma unroll
  for (int n = 0; n < KK; ++n) { const float e = expf(lg[n] - m); se += e; num += e * as[ti[n]]; }
  coeff[b * 256 + c] = num / se;
}

// ---------------------------------------------------------------------------
// K8: out[t,b,f]: f<512 -> coeff[b,f>>1]*vsum[2t+(f&1)]*(t<len[b]); else PE
// ---------------------------------------------------------------------------
__global__ __launch_bounds__(256) void k_out(const float* __restrict__ coeff,
                                             const float* __restrict__ vsum,
                                             const float* __restrict__ times,
                                             const int* __restrict__ lengths,
                                             float* __restrict__ out) {
  const size_t idx = (size_t)blockIdx.x * 256 + threadIdx.x;
  const size_t total = (size_t)TT * BB * 528;
  if (idx >= total) return;
  const int f = (int)(idx % 528);
  const size_t tb = idx / 528;
  const int b = (int)(tb % BB);
  const int t = (int)(tb / BB);
  float val;
  if (f < 512) {
    const int c = f >> 1;
    val = (t < lengths[b]) ? coeff[b * 256 + c] * vsum[2 * t + (f & 1)] : 0.f;
  } else {
    const int p = f - 512;
    const int pp = p & 7;
    const float ts = powf(256.f, (float)pp * (1.f / 7.f));
    const float arg = times[(size_t)t * BB + b] / ts;
    val = (p < 8) ? sinf(arg) : cosf(arg);
  }
  out[idx] = val;
}

// ---------------------------------------------------------------------------
extern "C" void kernel_launch(void* const* d_in, const int* in_sizes, int n_in,
                              void* d_out, int out_size, void* d_ws, size_t ws_size,
                              hipStream_t stream) {
  (void)in_sizes; (void)n_in; (void)out_size; (void)ws_size;
  const float* src    = (const float*)d_in[0];
  const float* statc  = (const float*)d_in[1];
  const float* times  = (const float*)d_in[2];
  const int*   lens   = (const int*)  d_in[3];
  const float* Wsi    = (const float*)d_in[4];
  const float* bsi    = (const float*)d_in[5];
  const float* Wc     = (const float*)d_in[6];
  const float* bc     = (const float*)d_in[7];
  const float* Wq     = (const float*)d_in[8];
  const float* Wk     = (const float*)d_in[9];
  const float* Wv     = (const float*)d_in[10];
  float* out = (float*)d_out;

  // workspace layout (~35 MB)
  char* w = (char*)d_ws;
  _Float16* X   = (_Float16*)w; w += (size_t)BB * DINP * DD * 2;   // 33.5 MB
  _Float16* Mt  = (_Float16*)w; w += (size_t)DD * DD * 2;          // 0.5 MB
  float* ksum   = (float*)w; w += DD * 4;
  float* vsum   = (float*)w; w += DD * 4;
  float* w2     = (float*)w; w += DD * 4;
  float* wsum   = (float*)w; w += DINP * 4;
  float* abuf   = (float*)w; w += (size_t)BB * DSTAT * 4;
  float* logZ   = (float*)w; w += (size_t)BB * DSTAT * 4;
  float* ybuf   = (float*)w; w += (size_t)BB * DINP * 4;
  float* sAtt   = (float*)w; w += (size_t)BB * DINP * 4;
  float* coeff  = (float*)w; w += (size_t)BB * DINP * 4;

  k_prep   <<<BB * 64, 256, 0, stream>>>(src, X);
  k_weights<<<1, 512, 0, stream>>>(Wq, Wk, Wv, Wc, ksum, vsum, wsum, w2);
  k_mt     <<<1024, 256, 0, stream>>>(Wq, Wk, Mt);
  k_abs    <<<BB, 256, 0, stream>>>(Wsi, bsi, statc, abuf);
  k_logz   <<<BB, 256, 0, stream>>>(abuf, wsum, bc, logZ);
  k_y      <<<BB, 256, 0, stream>>>(X, w2, ybuf);
  k_attn   <<<BB * 16, 32, 0, stream>>>(X, Mt, ybuf, sAtt);
  k_coeff  <<<BB, 256, 0, stream>>>(abuf, logZ, wsum, sAtt, coeff);
  {
    const size_t total = (size_t)TT * BB * 528;
    k_out  <<<(unsigned)((total + 255) / 256), 256, 0, stream>>>(coeff, vsum, times, lens, out);
  }
}
